// MLPPredictor_86320252715186
// MI455X (gfx1250) — matrix operand order repaired and checked by reading, not compiled, not run on hardware
//
#include <hip/hip_runtime.h>
#include <hip/hip_bf16.h>

typedef __attribute__((ext_vector_type(16))) _Float16 v16h;
typedef __attribute__((ext_vector_type(8)))  float    v8f;
typedef __attribute__((ext_vector_type(4)))  float    f4;

#define N_NODES 50000
#define N_EDGES 500000
#define DIMF    128
#define NCLS    50

// ---------------------------------------------------------------------------
// Packed B-fragment layout (16-bit B, 32x16 KxN per fragment, wave32):
//   lane l: n = nt*16 + (l&15), K-half by (l>>4); element j: k = kb*32 + 16*(j>>3) + 8*(l>>4) + (j&7)
//   storage: frag-major, 32 lanes * 16 f16 (32 bytes) per lane, contiguous.
// Wpq: KB=4 (K=128), NT=8 (N=128: cols 0..63 -> W1 block, 64..127 -> W2 block)
// W3 : KB=4 (K=128), NT=4 (N=64 : cols 0..49 valid)
//
// PQ table layout (fp32, [N_NODES][128]), swizzled for b128 gathers:
//   element (row, block, n, ntl)  at  row*128 + block*64 + n*4 + ntl
//   where block: 0=P(W1) 1=Q(W2), n = c&15, ntl = c>>4  (c = class column 0..63)
// ---------------------------------------------------------------------------

__global__ void pack_w_kernel(const float* __restrict__ W,
                              _Float16* __restrict__ wpq,
                              _Float16* __restrict__ w3) {
    const int total_pq = 4 * 8 * 512;   // frags * 512 f16 each
    const int total_w3 = 4 * 4 * 512;
    for (int idx = blockIdx.x * blockDim.x + threadIdx.x;
         idx < total_pq + total_w3; idx += gridDim.x * blockDim.x) {
        bool isPQ = idx < total_pq;
        int li    = isPQ ? idx : (idx - total_pq);
        int frag  = li >> 9;
        int lane  = (li >> 4) & 31;
        int j     = li & 15;
        int kb, nt;
        if (isPQ) { kb = frag >> 3; nt = frag & 7; }
        else      { kb = frag >> 2; nt = frag & 3; }
        int n = nt * 16 + (lane & 15);
        int k = kb * 32 + 16 * (j >> 3) + 8 * (lane >> 4) + (j & 7);
        float val = 0.0f;
        if (isPQ) {
            if (n < 64) {                       // W1 block: contribution of h[src]
                if (n < NCLS) val = W[(size_t)n * (3 * DIMF) + k];
            } else {                            // W2 block: contribution of h[dst]
                int c = n - 64;
                if (c < NCLS) val = W[(size_t)c * (3 * DIMF) + DIMF + k];
            }
            wpq[li] = (_Float16)val;
        } else {                                // W3 block: contribution of edge feats
            if (n < NCLS) val = W[(size_t)n * (3 * DIMF) + 2 * DIMF + k];
            w3[li] = (_Float16)val;
        }
    }
}

// Build a 16x32 f16 A-fragment for this lane from 4 float4 runs
// (16-bit A layout: element j -> K = 16*(j>=8) + 8*(lane>>4) + (j&7)).
__device__ inline v16h make_afrag(f4 f0, f4 f1, f4 f2, f4 f3) {
    v16h a;
    a[0]  = (_Float16)f0.x; a[1]  = (_Float16)f0.y; a[2]  = (_Float16)f0.z; a[3]  = (_Float16)f0.w;
    a[4]  = (_Float16)f1.x; a[5]  = (_Float16)f1.y; a[6]  = (_Float16)f1.z; a[7]  = (_Float16)f1.w;
    a[8]  = (_Float16)f2.x; a[9]  = (_Float16)f2.y; a[10] = (_Float16)f2.z; a[11] = (_Float16)f2.w;
    a[12] = (_Float16)f3.x; a[13] = (_Float16)f3.y; a[14] = (_Float16)f3.z; a[15] = (_Float16)f3.w;
    return a;
}

// ---------------------------------------------------------------------------
// Kernel 1: PQ = h @ [W1^T | W2^T] -> fp32 table [N_NODES][128] (swizzled)
// One wave per 16-node tile; 8 column tiles; 4 K-blocks; 32 WMMAs per tile.
// ---------------------------------------------------------------------------
__global__ void pq_gemm_kernel(const float* __restrict__ h,
                               const _Float16* __restrict__ wpq,
                               float* __restrict__ pq) {
    const int wave  = (blockIdx.x * blockDim.x + threadIdx.x) >> 5;
    const int lane  = threadIdx.x & 31;
    const int ntile = N_NODES / 16;       // 3125
    if (wave >= ntile) return;
    const int half = lane >> 4;
    const int n    = lane & 15;
    const float* hrow = h + (size_t)(wave * 16 + n) * DIMF;   // A row m = lane&15

    v8f acc[8];
    #pragma unroll
    for (int t = 0; t < 8; ++t) acc[t] = (v8f){0,0,0,0,0,0,0,0};

    #pragma unroll
    for (int kb = 0; kb < 4; ++kb) {
        const int k0 = kb * 32 + 8 * half;
        f4 f0 = *(const f4*)(hrow + k0);
        f4 f1 = *(const f4*)(hrow + k0 + 4);
        f4 f2 = *(const f4*)(hrow + k0 + 16);
        f4 f3 = *(const f4*)(hrow + k0 + 20);
        v16h a = make_afrag(f0, f1, f2, f3);
        #pragma unroll
        for (int nt = 0; nt < 8; ++nt) {
            v16h b = *(const v16h*)(wpq + ((size_t)(kb * 8 + nt) * 32 + lane) * 16);
            acc[nt] = __builtin_amdgcn_wmma_f32_16x16x32_f16(
                false, a, false, b, (short)0, acc[nt], false, false);
        }
    }
    // D layout: VGPR v -> m = v + 8*half ; class col c = nt*16 + n
    // Swizzled PQ store: row*128 + block*64 + n*4 + ntl
    #pragma unroll
    for (int nt = 0; nt < 8; ++nt) {
        int block = nt >> 2;
        int ntl   = nt & 3;
        #pragma unroll
        for (int v = 0; v < 8; ++v) {
            int m = v + 8 * half;
            pq[(size_t)(wave * 16 + m) * 128 + block * 64 + n * 4 + ntl] = acc[nt][v];
        }
    }
}

// ---------------------------------------------------------------------------
// Kernel 2: persistent waves, W3 B-fragments held in 128 VGPRs.
// Per 16-edge tile: acc = e_tile @ W3^T (16 back-to-back WMMAs), then
// out = acc + P[src] + Q[dst] + b with one b128 gather per P/Q row.
// ---------------------------------------------------------------------------
__global__ __launch_bounds__(256) void
edge_score_kernel(const float* __restrict__ efeat,
                  const int* __restrict__ src,
                  const int* __restrict__ dst,
                  const _Float16* __restrict__ w3,
                  const float* __restrict__ pq,
                  const float* __restrict__ bias,
                  float* __restrict__ out) {
    const int lane   = threadIdx.x & 31;
    const int gwave  = (blockIdx.x * blockDim.x + threadIdx.x) >> 5;
    const int nwaves = (gridDim.x * blockDim.x) >> 5;
    const int half   = lane >> 4;
    const int n      = lane & 15;
    const int etile  = N_EDGES / 16;      // 31250

    // Hoist all 16 B fragments into registers (128 VGPRs), once per wave.
    v16h Bf[16];
    #pragma unroll
    for (int f = 0; f < 16; ++f)
        Bf[f] = *(const v16h*)(w3 + ((size_t)f * 32 + lane) * 16);

    // Hoist bias for this lane's 4 class columns.
    float bv[4];
    #pragma unroll
    for (int nt = 0; nt < 4; ++nt) {
        int c = nt * 16 + n;
        bv[nt] = (c < NCLS) ? bias[c] : 0.0f;
    }

    for (int tile = gwave; tile < etile; tile += nwaves) {
        const int ebase = tile * 16;
        const float* erow = efeat + (size_t)(ebase + n) * DIMF;

        // Kick off the index loads early so the P/Q gather chain overlaps WMMAs.
        int sv[8], dv[8];
        #pragma unroll
        for (int v = 0; v < 8; ++v) {
            int e = ebase + v + 8 * half;
            sv[v] = src[e];
            dv[v] = dst[e];
        }

        v8f acc[4];
        #pragma unroll
        for (int t = 0; t < 4; ++t) acc[t] = (v8f){0,0,0,0,0,0,0,0};

        #pragma unroll
        for (int kb = 0; kb < 4; ++kb) {
            const int k0 = kb * 32 + 8 * half;
            f4 f0 = __builtin_nontemporal_load((const f4*)(erow + k0));
            f4 f1 = __builtin_nontemporal_load((const f4*)(erow + k0 + 4));
            f4 f2 = __builtin_nontemporal_load((const f4*)(erow + k0 + 16));
            f4 f3 = __builtin_nontemporal_load((const f4*)(erow + k0 + 20));
            v16h a = make_afrag(f0, f1, f2, f3);
            #pragma unroll
            for (int nt = 0; nt < 4; ++nt) {
                acc[nt] = __builtin_amdgcn_wmma_f32_16x16x32_f16(
                    false, a, false, Bf[kb * 4 + nt], (short)0, acc[nt], false, false);
            }
        }

        #pragma unroll
        for (int v = 0; v < 8; ++v) {
            int m = v + 8 * half;
            int e = ebase + m;
            // Swizzled PQ: this lane's 4 class columns are contiguous -> b128 gathers.
            f4 Pv = *(const f4*)(pq + (size_t)sv[v] * 128 + n * 4);
            f4 Qv = *(const f4*)(pq + (size_t)dv[v] * 128 + 64 + n * 4);
            float pvals[4] = {Pv.x, Pv.y, Pv.z, Pv.w};
            float qvals[4] = {Qv.x, Qv.y, Qv.z, Qv.w};
            #pragma unroll
            for (int nt = 0; nt < 4; ++nt) {
                int c = nt * 16 + n;
                if (c < NCLS) {
                    float val = acc[nt][v] + pvals[nt] + qvals[nt] + bv[nt];
                    __builtin_nontemporal_store(val, out + (size_t)e * NCLS + c);
                }
            }
        }
    }
}

// Scalar fallback if scratch is too small for the PQ table (not expected).
__global__ void fallback_kernel(const float* __restrict__ h, const float* __restrict__ efeat,
                                const int* __restrict__ src, const int* __restrict__ dst,
                                const float* __restrict__ W, const float* __restrict__ b,
                                float* __restrict__ out) {
    long long idx = (long long)blockIdx.x * blockDim.x + threadIdx.x;
    if (idx >= (long long)N_EDGES * NCLS) return;
    int edge = (int)(idx / NCLS);
    int c    = (int)(idx % NCLS);
    const float* hu = h + (size_t)src[edge] * DIMF;
    const float* hv = h + (size_t)dst[edge] * DIMF;
    const float* ee = efeat + (size_t)edge * DIMF;
    const float* w  = W + (size_t)c * (3 * DIMF);
    float s = b[c];
    for (int k = 0; k < DIMF; ++k)
        s += hu[k] * w[k] + hv[k] * w[DIMF + k] + ee[k] * w[2 * DIMF + k];
    out[idx] = s;
}

extern "C" void kernel_launch(void* const* d_in, const int* in_sizes, int n_in,
                              void* d_out, int out_size, void* d_ws, size_t ws_size,
                              hipStream_t stream) {
    const float* h    = (const float*)d_in[0];
    const float* e    = (const float*)d_in[1];
    const int*   src  = (const int*)d_in[2];
    const int*   dst  = (const int*)d_in[3];
    const float* W    = (const float*)d_in[4];
    const float* b    = (const float*)d_in[5];
    float*       out  = (float*)d_out;

    const size_t wpq_bytes = (size_t)4 * 8 * 512 * sizeof(_Float16);  // 32 KB
    const size_t pq_off    = 49152;                                   // 48 KB (aligned)
    const size_t pq_bytes  = (size_t)N_NODES * 128 * sizeof(float);   // 25.6 MB

    if (ws_size < pq_off + pq_bytes) {
        long long total = (long long)N_EDGES * NCLS;
        int blocks = (int)((total + 255) / 256);
        fallback_kernel<<<blocks, 256, 0, stream>>>(h, e, src, dst, W, b, out);
        return;
    }

    _Float16* wpq = (_Float16*)d_ws;
    _Float16* w3  = (_Float16*)((char*)d_ws + wpq_bytes);
    float*    pq  = (float*)((char*)d_ws + pq_off);

    pack_w_kernel<<<24, 256, 0, stream>>>(W, wpq, w3);

    {   // one wave per 16-node tile
        int waves  = N_NODES / 16;            // 3125
        int blocks = (waves * 32 + 255) / 256;
        pq_gemm_kernel<<<blocks, 256, 0, stream>>>(h, wpq, pq);
    }
    {   // persistent waves grid-striding over 31250 edge tiles
        edge_score_kernel<<<1024, 256, 0, stream>>>(e, src, dst, w3, pq, b, out);
    }
}
